// MultiModalVAE_46059229283223
// MI455X (gfx1250) — matrix operand order repaired and checked by reading, not compiled, not run on hardware
//
#include <hip/hip_runtime.h>

typedef _Float16 v16h __attribute__((ext_vector_type(16)));
typedef _Float16 v8h  __attribute__((ext_vector_type(8)));
typedef _Float16 v4h  __attribute__((ext_vector_type(4)));
typedef float    v8f  __attribute__((ext_vector_type(8)));

namespace mmv {

constexpr int CB = 4, CS = 1024, CD = 1024, CH = 8, CHD = 128, CM = 4;

// ---------------- fp32 -> fp16 conversion (vectorized) ----------------
__global__ void f32_to_f16(const float* __restrict__ in, _Float16* __restrict__ out, int n4) {
  int i = blockIdx.x * blockDim.x + threadIdx.x;
  if (i < n4) {
    float4 v = ((const float4*)in)[i];
    v4h o;
    o[0] = (_Float16)v.x; o[1] = (_Float16)v.y;
    o[2] = (_Float16)v.z; o[3] = (_Float16)v.w;
    ((v4h*)out)[i] = o;
  }
}

// ---------------- WMMA fragment loads ----------------
// A: 16x32 (MxK) f16, row-major, ld = K.
// ISA layout: lanes 0-15 hold row M=lane with K {0..7, 16..23};
//             lanes 16-31 hold row M=lane-16 with K {8..15, 24..31}.
__device__ __forceinline__ v16h load_fragA(const _Float16* __restrict__ base, int ld,
                                           int row0, int k0, int lane) {
  const int hl = lane >> 4, idx = lane & 15;
  const _Float16* p = base + (size_t)(row0 + idx) * (size_t)ld + k0;
  v8h a = *(const v8h*)(p + (hl ? 8 : 0));
  v8h b = *(const v8h*)(p + (hl ? 24 : 16));
  v16h r;
#pragma unroll
  for (int e = 0; e < 8; ++e) { r[e] = a[e]; r[e + 8] = b[e]; }
  return r;
}

// B: 32x16 (KxN) f16. We hold W row-major as N x K (i.e. compute A @ W^T), so
// B column n is W row n (contiguous over K).
// B layout (matches A): lanes 0-15 -> col N=lane, K {0..7, 16..23};
//                       lanes 16-31 -> col N=lane-16, K {8..15, 24..31}.
__device__ __forceinline__ v16h load_fragB(const _Float16* __restrict__ base, int ld,
                                           int n0, int k0, int lane) {
  const int hl = lane >> 4, idx = lane & 15;
  const _Float16* p = base + (size_t)(n0 + idx) * (size_t)ld + k0;
  v8h a = *(const v8h*)(p + (hl ? 8 : 0));
  v8h b = *(const v8h*)(p + (hl ? 24 : 16));
  v16h r;
#pragma unroll
  for (int e = 0; e < 8; ++e) { r[e] = a[e]; r[e + 8] = b[e]; }
  return r;
}

// C/D 16x16 f32: VGPR r -> M = r + (lane>=16 ? 8 : 0), N = lane & 15.
template<bool OUT_F16, bool REMAP>
__device__ __forceinline__ void store_tile(v8f acc, void* __restrict__ outp,
                                           const float* __restrict__ bias,
                                           const float* __restrict__ emb,
                                           int row0, int n, int N, int mod, int lane) {
  const int hl = lane >> 4;
  float bn = bias ? bias[n] : 0.0f;
  if (emb) bn += emb[n];
#pragma unroll
  for (int r = 0; r < 8; ++r) {
    int orow = row0 + r + (hl ? 8 : 0);
    if (REMAP) {
      // row (b*S + s) of modality `mod` -> stacked row ((b*M + mod)*S + s)
      int bb = orow >> 10;            // CS == 1024
      int ss = orow & (CS - 1);
      orow = ((bb * CM + mod) << 10) + ss;
    }
    float v = acc[r] + bn;
    if (OUT_F16) ((_Float16*)outp)[(size_t)orow * N + n] = (_Float16)v;
    else         ((float*)outp)[(size_t)orow * N + n] = v;
  }
}

// C[Mrows x N] = A[Mrows x K] @ W^T + bias (+ emb), f16 inputs, f32 accumulate.
// One wave -> 16x64 output strip (A fragment reused across 4 WMMAs).
template<bool OUT_F16, bool REMAP>
__global__ void gemm_wmma_f16(const _Float16* __restrict__ A, const _Float16* __restrict__ W,
                              const float* __restrict__ bias, const float* __restrict__ emb,
                              void* __restrict__ outp, int Mrows, int N, int K, int mod) {
  const int lane = threadIdx.x & 31;
  const int wave = threadIdx.x >> 5;
  const int tilesN = N >> 6;
  const int tile = blockIdx.x * (blockDim.x >> 5) + wave;
  const int row0 = (tile / tilesN) << 4;
  const int col0 = (tile % tilesN) << 6;
  if (row0 >= Mrows) return;   // wave-uniform: EXEC stays all-ones for WMMA

  v8f acc0 = {}, acc1 = {}, acc2 = {}, acc3 = {};
  for (int k0 = 0; k0 < K; k0 += 32) {
    __builtin_prefetch(A + (size_t)row0 * K + k0 + 256, 0, 1);   // global_prefetch_b8
    v16h a  = load_fragA(A, K, row0, k0, lane);
    v16h b0 = load_fragB(W, K, col0 +  0, k0, lane);
    v16h b1 = load_fragB(W, K, col0 + 16, k0, lane);
    v16h b2 = load_fragB(W, K, col0 + 32, k0, lane);
    v16h b3 = load_fragB(W, K, col0 + 48, k0, lane);
    acc0 = __builtin_amdgcn_wmma_f32_16x16x32_f16(false, a, false, b0, (short)0, acc0, false, false);
    acc1 = __builtin_amdgcn_wmma_f32_16x16x32_f16(false, a, false, b1, (short)0, acc1, false, false);
    acc2 = __builtin_amdgcn_wmma_f32_16x16x32_f16(false, a, false, b2, (short)0, acc2, false, false);
    acc3 = __builtin_amdgcn_wmma_f32_16x16x32_f16(false, a, false, b3, (short)0, acc3, false, false);
  }
  const int idx = lane & 15;
  store_tile<OUT_F16, REMAP>(acc0, outp, bias, emb, row0, col0 +  0 + idx, N, mod, lane);
  store_tile<OUT_F16, REMAP>(acc1, outp, bias, emb, row0, col0 + 16 + idx, N, mod, lane);
  store_tile<OUT_F16, REMAP>(acc2, outp, bias, emb, row0, col0 + 32 + idx, N, mod, lane);
  store_tile<OUT_F16, REMAP>(acc3, outp, bias, emb, row0, col0 + 48 + idx, N, mod, lane);
}

// ---------------- routed cross-modal attention + mean-fuse ----------------
// One wave per (b, s, h): WIN=3 softmax over modalities, dot over HD=128
// (4 f32 per lane, wave32 shfl_xor reduction). fused = mean over m.
__global__ void attn_route(const _Float16* __restrict__ Q, const _Float16* __restrict__ Kt,
                           const _Float16* __restrict__ Vt, const float* __restrict__ temp,
                           _Float16* __restrict__ fused) {
  const int lane = threadIdx.x & 31;
  const int wid = (blockIdx.x * blockDim.x + threadIdx.x) >> 5;
  const int h = wid & (CH - 1);
  const int s = (wid >> 3) & (CS - 1);
  const int b = wid >> 13;                       // / (CH*CS)
  const float scale = 1.0f / (sqrtf((float)CHD) * fabsf(temp[0]));
  // Static Cantor routing table (double-precision arithmetic of the reference)
  const int routes[CM][3] = {{0,1,2},{0,1,2},{2,3,0},{3,2,0}};
  const size_t seg = (size_t)h * CHD + (size_t)lane * 4;
  float a0 = 0.f, a1 = 0.f, a2 = 0.f, a3 = 0.f;
#pragma unroll
  for (int m = 0; m < CM; ++m) {
    const size_t qoff = ((size_t)((b * CM + m) * CS + s)) * CD + seg;
    v4h qh = *(const v4h*)(Q + qoff);
    float sc[3];
#pragma unroll
    for (int w = 0; w < 3; ++w) {
      const size_t koff = ((size_t)((b * CM + routes[m][w]) * CS + s)) * CD + seg;
      v4h kh = *(const v4h*)(Kt + koff);
      float p = (float)qh[0] * (float)kh[0] + (float)qh[1] * (float)kh[1]
              + (float)qh[2] * (float)kh[2] + (float)qh[3] * (float)kh[3];
#pragma unroll
      for (int o = 16; o > 0; o >>= 1) p += __shfl_xor(p, o, 32);
      sc[w] = p * scale;
    }
    float mx = fmaxf(sc[0], fmaxf(sc[1], sc[2]));
    float e0 = __expf(sc[0] - mx), e1 = __expf(sc[1] - mx), e2 = __expf(sc[2] - mx);
    float inv = 1.0f / (e0 + e1 + e2);
    float aw[3] = {e0 * inv, e1 * inv, e2 * inv};
#pragma unroll
    for (int w = 0; w < 3; ++w) {
      const size_t voff = ((size_t)((b * CM + routes[m][w]) * CS + s)) * CD + seg;
      v4h vh = *(const v4h*)(Vt + voff);
      a0 += aw[w] * (float)vh[0]; a1 += aw[w] * (float)vh[1];
      a2 += aw[w] * (float)vh[2]; a3 += aw[w] * (float)vh[3];
    }
  }
  v4h o;
  o[0] = (_Float16)(a0 * 0.25f); o[1] = (_Float16)(a1 * 0.25f);
  o[2] = (_Float16)(a2 * 0.25f); o[3] = (_Float16)(a3 * 0.25f);
  *(v4h*)(fused + ((size_t)(b * CS + s)) * CD + seg) = o;
}

} // namespace mmv

extern "C" void kernel_launch(void* const* d_in, const int* in_sizes, int n_in,
                              void* d_out, int out_size, void* d_ws, size_t ws_size,
                              hipStream_t stream) {
  using namespace mmv;
  (void)in_sizes; (void)n_in; (void)out_size; (void)ws_size;

  const float* x[4]  = {(const float*)d_in[0], (const float*)d_in[3], (const float*)d_in[6], (const float*)d_in[9]};
  const float* Wm[4] = {(const float*)d_in[1], (const float*)d_in[4], (const float*)d_in[7], (const float*)d_in[10]};
  const float* bm[4] = {(const float*)d_in[2], (const float*)d_in[5], (const float*)d_in[8], (const float*)d_in[11]};
  const float* emb = (const float*)d_in[12];
  const float* Wq = (const float*)d_in[13]; const float* bq = (const float*)d_in[14];
  const float* Wk = (const float*)d_in[15]; const float* bk = (const float*)d_in[16];
  const float* Wv = (const float*)d_in[17]; const float* bv = (const float*)d_in[18];
  const float* Wo = (const float*)d_in[19]; const float* bo = (const float*)d_in[20];
  const float* temperature = (const float*)d_in[21];
  const int din[4] = {768, 1024, 512, 1024};

  // Workspace carve-up (all f16): ~185 MB total.
  char* ws = (char*)d_ws;
  size_t off = 0;
  auto wsAlloc = [&](size_t elems) {
    _Float16* p = (_Float16*)(ws + off);
    off += (elems * sizeof(_Float16) + 255) & ~(size_t)255;
    return p;
  };
  _Float16 *x16[4], *wm16[4];
  for (int m = 0; m < 4; ++m) x16[m]  = wsAlloc((size_t)CB * CS * din[m]);
  for (int m = 0; m < 4; ++m) wm16[m] = wsAlloc((size_t)CD * din[m]);
  _Float16* wq16 = wsAlloc((size_t)CD * CD);
  _Float16* wk16 = wsAlloc((size_t)CD * CD);
  _Float16* wv16 = wsAlloc((size_t)CD * CD);
  _Float16* wo16 = wsAlloc((size_t)CD * CD);
  _Float16* st16 = wsAlloc((size_t)CB * CM * CS * CD);   // stacked
  _Float16* q16  = wsAlloc((size_t)CB * CM * CS * CD);
  _Float16* k16  = wsAlloc((size_t)CB * CM * CS * CD);
  _Float16* v16  = wsAlloc((size_t)CB * CM * CS * CD);
  _Float16* f16t = wsAlloc((size_t)CB * CS * CD);        // fused

  auto conv = [&](const float* src, _Float16* dst, size_t n) {
    int n4 = (int)(n >> 2);
    f32_to_f16<<<(n4 + 255) / 256, 256, 0, stream>>>(src, dst, n4);
  };
  for (int m = 0; m < 4; ++m) conv(x[m],  x16[m],  (size_t)CB * CS * din[m]);
  for (int m = 0; m < 4; ++m) conv(Wm[m], wm16[m], (size_t)CD * din[m]);
  conv(Wq, wq16, (size_t)CD * CD);
  conv(Wk, wk16, (size_t)CD * CD);
  conv(Wv, wv16, (size_t)CD * CD);
  conv(Wo, wo16, (size_t)CD * CD);

  // 1) Input projections -> stacked (f16), with bias + emb[m], row-remapped.
  for (int m = 0; m < 4; ++m) {
    int Mrows = CB * CS;
    int tiles = (Mrows / 16) * (CD / 64);
    gemm_wmma_f16<true, true><<<tiles / 4, 128, 0, stream>>>(
        x16[m], wm16[m], bm[m], emb + (size_t)m * CD, st16, Mrows, CD, din[m], m);
  }
  // 2) Q/K/V GEMMs (16384 x 1024 x 1024 each).
  {
    int Mrows = CB * CM * CS;
    int tiles = (Mrows / 16) * (CD / 64);
    gemm_wmma_f16<true, false><<<tiles / 4, 128, 0, stream>>>(st16, wq16, bq, nullptr, q16, Mrows, CD, CD, 0);
    gemm_wmma_f16<true, false><<<tiles / 4, 128, 0, stream>>>(st16, wk16, bk, nullptr, k16, Mrows, CD, CD, 0);
    gemm_wmma_f16<true, false><<<tiles / 4, 128, 0, stream>>>(st16, wv16, bv, nullptr, v16, Mrows, CD, CD, 0);
  }
  // 3) Routed attention + mean over modalities -> fused (f16).
  {
    int waves = CB * CS * CH;                    // 32768
    attn_route<<<(waves * 32) / 256, 256, 0, stream>>>(q16, k16, v16, temperature, f16t);
  }
  // 4) Output projection -> d_out (f32).
  {
    int Mrows = CB * CS;
    int tiles = (Mrows / 16) * (CD / 64);
    gemm_wmma_f16<false, false><<<tiles / 4, 128, 0, stream>>>(f16t, wo16, bo, nullptr, d_out, Mrows, CD, CD, 0);
  }
}
